// GatEncoder2_32238024524375
// MI455X (gfx1250) — matrix operand order repaired and checked by reading, not compiled, not run on hardware
//
#include <hip/hip_runtime.h>
#include <math.h>

// ---------------- problem constants (match reference) ----------------
#define Nn    50000
#define Ee    800000
#define Etot  (Ee + Nn)        // edges + self loops = 850000
#define FIN   32
#define Hh    4
#define Cc    64
#define HIDN  256              // Hh * Cc
#define Gg    1024
#define MLPH  2048
#define NOUTc 768
#define SLOPE 0.2f

typedef __attribute__((ext_vector_type(16))) _Float16 v16h;
typedef __attribute__((ext_vector_type(8)))  float    v8f;

#ifndef __has_builtin
#define __has_builtin(x) 0
#endif

#define SHARED_AS __attribute__((address_space(3)))

// Use CDNA5 async global->LDS DMA (ASYNCcnt-tracked) via inline asm.
#define USE_ASYNC_LDS 1

#if __has_builtin(__builtin_amdgcn_s_wait_asynccnt)
#define WAIT_ASYNC() __builtin_amdgcn_s_wait_asynccnt(0)
#else
#define WAIT_ASYNC() asm volatile("s_wait_asynccnt 0" ::: "memory")
#endif

__device__ __forceinline__ void async_copy_b128(const void* gptr, void* lptr) {
    // VDST = per-lane LDS byte offset; VADDR = 64-bit global address
    unsigned lds = (unsigned)(unsigned long long)(SHARED_AS void*)lptr;
    unsigned long long ga = (unsigned long long)gptr;
    asm volatile("global_load_async_to_lds_b128 %0, %1, off"
                 :: "v"(lds), "v"(ga) : "memory");
}

// ---------------- small utility kernels ----------------
__global__ void k_fill_f32(float* __restrict__ p, float v, int n) {
    int t = blockIdx.x * blockDim.x + threadIdx.x;
    if (t < n) p[t] = v;
}

__global__ void k_cast_f16(const float* __restrict__ in, _Float16* __restrict__ out, int n) {
    int t = blockIdx.x * blockDim.x + threadIdx.x;
    if (t < n) out[t] = (_Float16)in[t];
}

__global__ void k_relu_cast_f16(const float* __restrict__ in, _Float16* __restrict__ out, int n) {
    int t = blockIdx.x * blockDim.x + threadIdx.x;
    if (t < n) out[t] = (_Float16)fmaxf(in[t], 0.0f);
}

__global__ void k_init_bias(const float* __restrict__ b, float* __restrict__ out, int n) {
    int t = blockIdx.x * blockDim.x + threadIdx.x;      // n = Nn*HIDN
    if (t < n) out[t] = b[t & (HIDN - 1)];
}

// ---------------- WMMA f16 GEMM: C[M,Nc] = A[M,K] * B[K,Nc] (+bias)(+relu) ----------------
// Block: 256 threads = 8 waves. Block tile 64(M) x 128(N), K staged 32 wide.
// Wave (wm,wn) = (wave&1, wave>>1) owns a 32x32 patch = 2x2 WMMA tiles -> 4 wmmas/K-step
// with A-fragments reused across N-subtiles and B-fragments across M-subtiles.
#define LDA_S 40    // A LDS stride in halves (80B rows)
#define LDB_S 136   // B LDS stride in halves (272B rows)

__global__ __launch_bounds__(256)
void k_gemm_f16(const _Float16* __restrict__ A, const _Float16* __restrict__ B,
                float* __restrict__ C, const float* __restrict__ bias,
                int M, int K, int Ncol, int relu)
{
    __shared__ _Float16 As[64 * LDA_S];   // 64 x 32 tile
    __shared__ _Float16 Bs[32 * LDB_S];   // 32 x 128 tile

    const int tid  = threadIdx.x;
    const int wave = tid >> 5;
    const int lane = tid & 31;
    const int g    = lane >> 4;      // half-wave group
    const int ml   = lane & 15;
    const int wm   = wave & 1;       // 0..1 : 32-row group
    const int wn   = wave >> 1;      // 0..3 : 32-col group

    const int bm = (int)blockIdx.x * 64;
    const int bn = (int)blockIdx.y * 128;

    // staging coordinates
    const int ar = tid >> 2;             // 0..63 A row
    const int ac = (tid & 3) * 8;        // 16B chunk in A row
    const int br = tid >> 4;             // 0..15 B row (and +16 for second chunk)
    const int bc = (tid & 15) * 8;       // 16B chunk in B row

    v8f acc[2][2] = {};

    for (int k0 = 0; k0 < K; k0 += 32) {
#if USE_ASYNC_LDS
        // ---- async DMA global -> LDS (ASYNCcnt-tracked) ----
        {
            int row = bm + ar;
            if (row < M) {
                async_copy_b128(A + (size_t)row * K + k0 + ac,
                                &As[ar * LDA_S + ac]);
            } else {
                *(float4*)&As[ar * LDA_S + ac] = make_float4(0.f, 0.f, 0.f, 0.f);
            }
            async_copy_b128(B + (size_t)(k0 + br) * Ncol + bn + bc,
                            &Bs[br * LDB_S + bc]);
            async_copy_b128(B + (size_t)(k0 + br + 16) * Ncol + bn + bc,
                            &Bs[(br + 16) * LDB_S + bc]);
        }
        WAIT_ASYNC();
#else
        // ---- staged through VGPRs ----
        {
            float4 va = make_float4(0.f, 0.f, 0.f, 0.f);
            int row = bm + ar;
            if (row < M)
                va = *(const float4*)(A + (size_t)row * K + k0 + ac);
            *(float4*)&As[ar * LDA_S + ac] = va;
        }
        *(float4*)&Bs[br * LDB_S + bc] =
            *(const float4*)(B + (size_t)(k0 + br) * Ncol + bn + bc);
        *(float4*)&Bs[(br + 16) * LDB_S + bc] =
            *(const float4*)(B + (size_t)(k0 + br + 16) * Ncol + bn + bc);
#endif
        // prefetch next K tile of B (global_prefetch_b8)
        if (k0 + 32 < K)
            __builtin_prefetch(B + (size_t)(k0 + 32 + br) * Ncol + bn + bc, 0, 1);

        __syncthreads();

        // ---- fragments per CDNA5 16-bit WMMA layouts (ISA 7.12.2) ----
        v16h af[2], bf[2];
        #pragma unroll
        for (int sm = 0; sm < 2; ++sm) {
            const _Float16* Arow = &As[(wm * 32 + sm * 16 + ml) * LDA_S];
            #pragma unroll
            for (int e = 0; e < 16; ++e) {
                int v = e >> 1, p = e & 1;
                af[sm][e] = Arow[((v & 4) << 2) + g * 8 + (v & 3) * 2 + p];
            }
        }
        #pragma unroll
        for (int sn = 0; sn < 2; ++sn) {
            const _Float16* Brow = &Bs[wn * 32 + sn * 16 + ml];
            #pragma unroll
            for (int e = 0; e < 16; ++e) bf[sn][e] = Brow[(((e & 8) << 1) + g * 8 + (e & 7)) * LDB_S];
        }
        #pragma unroll
        for (int sm = 0; sm < 2; ++sm)
            #pragma unroll
            for (int sn = 0; sn < 2; ++sn)
                acc[sm][sn] = __builtin_amdgcn_wmma_f32_16x16x32_f16(
                    false, af[sm], false, bf[sn], (short)0, acc[sm][sn], false, false);

        __syncthreads();
    }

    // ---- store (C/D layout: VGPR r -> M = 8*g + r, N = ml) ----
    #pragma unroll
    for (int sm = 0; sm < 2; ++sm) {
        #pragma unroll
        for (int sn = 0; sn < 2; ++sn) {
            int col = bn + wn * 32 + sn * 16 + ml;
            #pragma unroll
            for (int r = 0; r < 8; ++r) {
                int row = bm + wm * 32 + sm * 16 + g * 8 + r;
                if (row < M) {
                    float v = acc[sm][sn][r];
                    if (bias) v += bias[col];
                    if (relu) v = fmaxf(v, 0.0f);
                    C[(size_t)row * Ncol + col] = v;
                }
            }
        }
    }
}

// ---------------- attention ----------------
__global__ void k_att_logits(const float* __restrict__ h, const float* __restrict__ a_s,
                             const float* __restrict__ a_d,
                             float* __restrict__ ls, float* __restrict__ ld)
{
    int t = blockIdx.x * blockDim.x + threadIdx.x;   // Nn*Hh threads
    if (t >= Nn * Hh) return;
    int n = t >> 2, hd = t & 3;
    const float* hp = h + (size_t)n * HIDN + hd * Cc;
    const float* as = a_s + hd * Cc;
    const float* ad = a_d + hd * Cc;
    float s = 0.f, d = 0.f;
    #pragma unroll 8
    for (int c = 0; c < Cc; ++c) { float v = hp[c]; s = fmaf(v, as[c], s); d = fmaf(v, ad[c], d); }
    ls[t] = s; ld[t] = d;
}

__device__ __forceinline__ int edge_src(const int* ei, int e) { return e < Ee ? ei[e]      : e - Ee; }
__device__ __forceinline__ int edge_dst(const int* ei, int e) { return e < Ee ? ei[Ee + e] : e - Ee; }

__device__ __forceinline__ void atomicMaxF(float* addr, float v) {
    // ordered-int trick; valid with -inf init
    if (v >= 0.f) atomicMax((int*)addr, __float_as_int(v));
    else          atomicMin((unsigned int*)addr, __float_as_uint(v));
}

__global__ void k_edge_max(const int* __restrict__ ei, const float* __restrict__ ls,
                           const float* __restrict__ ld, float* __restrict__ mx)
{
    int t = blockIdx.x * blockDim.x + threadIdx.x;   // Etot*Hh threads
    if (t >= Etot * Hh) return;
    int e = t >> 2, hd = t & 3;
    int s = edge_src(ei, e), d = edge_dst(ei, e);
    float l = ls[s * Hh + hd] + ld[d * Hh + hd];
    l = l > 0.f ? l : l * SLOPE;
    atomicMaxF(&mx[d * Hh + hd], l);
}

__global__ void k_fix_max(float* __restrict__ mx, int n) {
    int t = blockIdx.x * blockDim.x + threadIdx.x;
    if (t < n) { float v = mx[t]; if (!__builtin_isfinite(v)) mx[t] = 0.f; }
}

__global__ void k_edge_expsum(const int* __restrict__ ei, const float* __restrict__ ls,
                              const float* __restrict__ ld, const float* __restrict__ mx,
                              float* __restrict__ ealpha, float* __restrict__ denom)
{
    int t = blockIdx.x * blockDim.x + threadIdx.x;   // Etot*Hh threads
    if (t >= Etot * Hh) return;
    int e = t >> 2, hd = t & 3;
    int s = edge_src(ei, e), d = edge_dst(ei, e);
    float l = ls[s * Hh + hd] + ld[d * Hh + hd];
    l = l > 0.f ? l : l * SLOPE;
    float ex = expf(l - mx[d * Hh + hd]);
    ealpha[t] = ex;
    atomicAdd(&denom[d * Hh + hd], ex);
}

__global__ void k_edge_agg(const int* __restrict__ ei, const float* __restrict__ hsrc,
                           const float* __restrict__ ealpha, const float* __restrict__ denom,
                           float* __restrict__ out)
{
    int t = blockIdx.x * blockDim.x + threadIdx.x;   // Etot*64 threads (4 chans each)
    if (t >= Etot * 64) return;
    int e = t >> 6, c4 = t & 63;
    int s = edge_src(ei, e), d = edge_dst(ei, e);
    int hd = c4 >> 4;
    float alpha = ealpha[e * Hh + hd] / (denom[d * Hh + hd] + 1e-16f);
    float4 hv = *(const float4*)(hsrc + (size_t)s * HIDN + c4 * 4);
    float* op = out + (size_t)d * HIDN + c4 * 4;
    atomicAdd(op + 0, hv.x * alpha);
    atomicAdd(op + 1, hv.y * alpha);
    atomicAdd(op + 2, hv.z * alpha);
    atomicAdd(op + 3, hv.w * alpha);
}

// ---------------- pooling ----------------
__global__ void k_pool_sum(const float* __restrict__ h, const int* __restrict__ batch,
                           float* __restrict__ sums, float* __restrict__ counts)
{
    int t = blockIdx.x * blockDim.x + threadIdx.x;   // Nn*64 threads
    if (t >= Nn * 64) return;
    int n = t >> 6, c4 = t & 63;
    int b = batch[n];
    float4 hv = *(const float4*)(h + (size_t)n * HIDN + c4 * 4);
    float* sp = sums + (size_t)b * HIDN + c4 * 4;
    atomicAdd(sp + 0, hv.x);
    atomicAdd(sp + 1, hv.y);
    atomicAdd(sp + 2, hv.z);
    atomicAdd(sp + 3, hv.w);
    if (c4 == 0) atomicAdd(&counts[b], 1.0f);
}

__global__ void k_pool_finish(const float* __restrict__ sums, const float* __restrict__ counts,
                              _Float16* __restrict__ pooled16)
{
    int t = blockIdx.x * blockDim.x + threadIdx.x;   // Gg*HIDN
    if (t >= Gg * HIDN) return;
    float c = fmaxf(counts[t / HIDN], 1.0f);
    pooled16[t] = (_Float16)(sums[t] / c);
}

// ---------------- layernorm over last dim (wave32-aware) ----------------
__global__ __launch_bounds__(256)
void k_layernorm(const float* __restrict__ z, const float* __restrict__ gamma,
                 const float* __restrict__ beta, float* __restrict__ out)
{
    __shared__ float red0[8], red1[8];
    int g = blockIdx.x, tid = threadIdx.x;
    const float* zr = z + (size_t)g * NOUTc;
    float s = 0.f, ss = 0.f;
    for (int j = tid; j < NOUTc; j += 256) { float v = zr[j]; s += v; ss = fmaf(v, v, ss); }
    #pragma unroll
    for (int o = 16; o > 0; o >>= 1) {
        s  += __shfl_down(s,  o, 32);
        ss += __shfl_down(ss, o, 32);
    }
    int wv = tid >> 5, ln = tid & 31;
    if (ln == 0) { red0[wv] = s; red1[wv] = ss; }
    __syncthreads();
    if (tid == 0) {
        float a = 0.f, b = 0.f;
        #pragma unroll
        for (int i = 0; i < 8; ++i) { a += red0[i]; b += red1[i]; }
        red0[0] = a; red1[0] = b;
    }
    __syncthreads();
    float mu  = red0[0] * (1.0f / NOUTc);
    float var = red1[0] * (1.0f / NOUTc) - mu * mu;
    float inv = rsqrtf(var + 1e-5f);
    float* orow = out + (size_t)g * NOUTc;
    for (int j = tid; j < NOUTc; j += 256)
        orow[j] = gamma[j] * (zr[j] - mu) * inv + beta[j];
}

// ---------------- host orchestration ----------------
extern "C" void kernel_launch(void* const* d_in, const int* in_sizes, int n_in,
                              void* d_out, int out_size, void* d_ws, size_t ws_size,
                              hipStream_t stream)
{
    (void)in_sizes; (void)n_in; (void)out_size; (void)ws_size;

    const float* x     = (const float*)d_in[0];
    const int*   ei    = (const int*)d_in[1];
    const int*   batch = (const int*)d_in[2];
    const float* W[3]    = {(const float*)d_in[3], (const float*)d_in[7],  (const float*)d_in[11]};
    const float* a_s[3]  = {(const float*)d_in[4], (const float*)d_in[8],  (const float*)d_in[12]};
    const float* a_d[3]  = {(const float*)d_in[5], (const float*)d_in[9],  (const float*)d_in[13]};
    const float* bvec[3] = {(const float*)d_in[6], (const float*)d_in[10], (const float*)d_in[14]};
    const float* Wm1 = (const float*)d_in[15];
    const float* bm1 = (const float*)d_in[16];
    const float* Wm2 = (const float*)d_in[17];
    const float* bm2 = (const float*)d_in[18];
    const float* gamma = (const float*)d_in[19];
    const float* beta  = (const float*)d_in[20];
    float* outp = (float*)d_out;

    // carve workspace
    char* ws = (char*)d_ws;
    size_t off = 0;
    auto carve = [&](size_t bytes) -> void* {
        void* p = ws + off;
        off += (bytes + 255) & ~(size_t)255;
        return p;
    };
    _Float16* x16    = (_Float16*)carve((size_t)Nn * FIN * 2);
    _Float16* A16    = (_Float16*)carve((size_t)Nn * HIDN * 2);
    float*    hW     = (float*)   carve((size_t)Nn * HIDN * 4);
    float*    agg    = (float*)   carve((size_t)Nn * HIDN * 4);
    float*    ls     = (float*)   carve((size_t)Nn * Hh * 4);
    float*    ld     = (float*)   carve((size_t)Nn * Hh * 4);
    float*    mx     = (float*)   carve((size_t)Nn * Hh * 4);
    float*    denom  = (float*)   carve((size_t)Nn * Hh * 4);
    float*    ealpha = (float*)   carve((size_t)Etot * Hh * 4);
    _Float16* wbuf16 = (_Float16*)carve((size_t)MLPH * NOUTc * 2);   // largest weight
    float*    sums   = (float*)   carve((size_t)Gg * HIDN * 4);
    float*    counts = (float*)   carve((size_t)Gg * 4);
    _Float16* pool16 = (_Float16*)carve((size_t)Gg * HIDN * 2);
    float*    z1     = (float*)   carve((size_t)Gg * MLPH * 4);
    _Float16* z1_16  = (_Float16*)carve((size_t)Gg * MLPH * 2);
    float*    zf     = (float*)   carve((size_t)Gg * NOUTc * 4);

    const int BT = 256;
    auto blocks = [&](long long n) { return dim3((unsigned)((n + BT - 1) / BT)); };
    auto gemm = [&](const _Float16* A, const _Float16* Bm, float* Cm, const float* bias,
                    int M, int K, int Ncol, int relu) {
        dim3 grid((M + 63) / 64, Ncol / 128);
        k_gemm_f16<<<grid, 256, 0, stream>>>(A, Bm, Cm, bias, M, K, Ncol, relu);
    };

    // x -> f16
    k_cast_f16<<<blocks((long long)Nn * FIN), BT, 0, stream>>>(x, x16, Nn * FIN);

    // ---- 3 GAT layers ----
    for (int l = 0; l < 3; ++l) {
        int Kin = (l == 0) ? FIN : HIDN;
        const _Float16* Ain = (l == 0) ? x16 : A16;

        // weights -> f16, then h' = A @ W  (WMMA)
        k_cast_f16<<<blocks((long long)Kin * HIDN), BT, 0, stream>>>(W[l], wbuf16, Kin * HIDN);
        gemm(Ain, wbuf16, hW, nullptr, Nn, Kin, HIDN, 0);

        // attention logits per node/head
        k_att_logits<<<blocks((long long)Nn * Hh), BT, 0, stream>>>(hW, a_s[l], a_d[l], ls, ld);

        // segment softmax over incoming edges
        k_fill_f32<<<blocks((long long)Nn * Hh), BT, 0, stream>>>(mx, -INFINITY, Nn * Hh);
        k_edge_max<<<blocks((long long)Etot * Hh), BT, 0, stream>>>(ei, ls, ld, mx);
        k_fix_max<<<blocks((long long)Nn * Hh), BT, 0, stream>>>(mx, Nn * Hh);
        k_fill_f32<<<blocks((long long)Nn * Hh), BT, 0, stream>>>(denom, 0.f, Nn * Hh);
        k_edge_expsum<<<blocks((long long)Etot * Hh), BT, 0, stream>>>(ei, ls, ld, mx, ealpha, denom);

        // aggregate: out = b + sum_e alpha_e * h'[src]
        k_init_bias<<<blocks((long long)Nn * HIDN), BT, 0, stream>>>(bvec[l], agg, Nn * HIDN);
        k_edge_agg<<<blocks((long long)Etot * 64), BT, 0, stream>>>(ei, hW, ealpha, denom, agg);

        if (l < 2)   // relu + cast to next layer's f16 A matrix
            k_relu_cast_f16<<<blocks((long long)Nn * HIDN), BT, 0, stream>>>(agg, A16, Nn * HIDN);
    }

    // ---- global mean pool ----
    k_fill_f32<<<blocks((long long)Gg * HIDN), BT, 0, stream>>>(sums, 0.f, Gg * HIDN);
    k_fill_f32<<<blocks((long long)Gg), BT, 0, stream>>>(counts, 0.f, Gg);
    k_pool_sum<<<blocks((long long)Nn * 64), BT, 0, stream>>>(agg, batch, sums, counts);
    k_pool_finish<<<blocks((long long)Gg * HIDN), BT, 0, stream>>>(sums, counts, pool16);

    // ---- MLP head (WMMA, fused bias/relu) ----
    k_cast_f16<<<blocks((long long)HIDN * MLPH), BT, 0, stream>>>(Wm1, wbuf16, HIDN * MLPH);
    gemm(pool16, wbuf16, z1, bm1, Gg, HIDN, MLPH, 1);
    k_cast_f16<<<blocks((long long)Gg * MLPH), BT, 0, stream>>>(z1, z1_16, Gg * MLPH);
    k_cast_f16<<<blocks((long long)MLPH * NOUTc), BT, 0, stream>>>(Wm2, wbuf16, MLPH * NOUTc);
    gemm(z1_16, wbuf16, zf, bm2, Gg, MLPH, NOUTc, 0);

    // ---- layernorm -> output ----
    k_layernorm<<<dim3(Gg), 256, 0, stream>>>(zf, gamma, beta, outp);
}
